// SAGERegressor_36610301231499
// MI455X (gfx1250) — matrix operand order repaired and checked by reading, not compiled, not run on hardware
//
#include <hip/hip_runtime.h>
#include <hip/hip_bf16.h>
#include <math.h>

// ---------- WMMA types (gfx1250, wave32) ----------
typedef __attribute__((ext_vector_type(16))) __bf16          v16bf;
typedef __attribute__((ext_vector_type(16))) unsigned short  v16u;
typedef __attribute__((ext_vector_type(8)))  float           v8f;

__device__ __forceinline__ unsigned short f32_to_bf16(float f) {
  unsigned int u = __builtin_bit_cast(unsigned int, f);
  unsigned int r = u + 0x7FFFu + ((u >> 16) & 1u);   // round-to-nearest-even
  return (unsigned short)(r >> 16);
}

// ---------- utility kernels ----------
__global__ void zero_f4_kernel(float* __restrict__ p, long n4) {
  long i = (long)blockIdx.x * blockDim.x + threadIdx.x;
  if (i < n4) ((float4*)p)[i] = make_float4(0.f, 0.f, 0.f, 0.f);
}

__global__ void deg_accum_kernel(const int* __restrict__ dst, float* __restrict__ deg, int nE) {
  int e = blockIdx.x * blockDim.x + threadIdx.x;
  if (e < nE) atomicAdd(&deg[dst[e]], 1.0f);
}

__global__ void invdeg_kernel(float* __restrict__ deg, int n) {
  int i = blockIdx.x * blockDim.x + threadIdx.x;
  if (i < n) {
    float d = deg[i];
    deg[i] = 1.0f / fmaxf(d, 1.0f);
  }
}

// wave-per-edge scatter-add: each lane moves 2 float4 chunks of the 256-f32 row
__global__ void scatter_add_kernel(const float* __restrict__ h, const int* __restrict__ src,
                                   const int* __restrict__ dst, float* __restrict__ agg, int nE) {
  int gid  = blockIdx.x * blockDim.x + threadIdx.x;
  int e    = gid >> 5;
  int lane = gid & 31;
  if (e >= nE) return;
  int s = src[e], d = dst[e];
  const float4* hs = (const float4*)(h + (long)s * 256);
  float*        ad = agg + (long)d * 256;
#pragma unroll
  for (int c = 0; c < 2; ++c) {
    int chunk = lane + c * 32;           // 64 float4 chunks per row
    float4 v = hs[chunk];
    atomicAdd(ad + chunk * 4 + 0, v.x);
    atomicAdd(ad + chunk * 4 + 1, v.y);
    atomicAdd(ad + chunk * 4 + 2, v.z);
    atomicAdd(ad + chunk * 4 + 3, v.w);
  }
}

__global__ void scale_rows_kernel(float* __restrict__ agg, const float* __restrict__ invdeg, long n4) {
  long i = (long)blockIdx.x * blockDim.x + threadIdx.x;
  if (i >= n4) return;
  long row = i >> 6;                     // 64 float4 per 256-f32 row
  float s = invdeg[row];
  float4 v = ((float4*)agg)[i];
  v.x *= s; v.y *= s; v.z *= s; v.w *= s;
  ((float4*)agg)[i] = v;
}

// Pack 256x256 f32 weights into per-lane WMMA bf16 B-fragments.
// B (32x16, K x N) for V_WMMA_*_16X16X32_BF16:
//   lane = khalf*16 + n ; element e -> K = kt*32 + khalf*8 + (e&7) + (e>>3)*16, N = ct*16 + n
// Packed index: [mat][ct][kt][lane][e] , 2*16*8*32*16 = 131072 ushorts.
__global__ void pack_w_kernel(const float* __restrict__ W0, const float* __restrict__ W1,
                              unsigned short* __restrict__ pk) {
  int i = blockIdx.x * blockDim.x + threadIdx.x;
  if (i >= 2 * 65536) return;
  const float* W = (i & 65536) ? W1 : W0;
  int r    = i & 65535;
  int e    = r & 15;
  int lane = (r >> 4) & 31;
  int kt   = (r >> 9) & 7;
  int ct   = (r >> 12) & 15;
  int col  = ct * 16 + (lane & 15);
  int k    = kt * 32 + ((lane >> 4) << 3) + (e & 7) + ((e >> 3) << 4);
  pk[i] = f32_to_bf16(W[k * 256 + col]);
}

// Pack f32 activations [N,256] into WMMA A-fragment order (bf16), optionally
// scaling each row by rowscale[row] (fused mean-aggregate divide).
// A (16x32 MxK, 16-bit) layout: lane = hl*16+m holds
//   e=0..7  -> K = kt*32 + hl*8 + e
//   e=8..15 -> K = kt*32 + 16 + hl*8 + (e-8)
// Packed index: [(t*8 + kt)*32 + lane]*16 + e  == gid*16 (lane fastest).
__global__ void pack_a_kernel(const float* __restrict__ A, const float* __restrict__ rowscale,
                              unsigned short* __restrict__ pk, int nNodes) {
  int gid = blockIdx.x * blockDim.x + threadIdx.x;   // (t, kt, lane)
  int total = (nNodes >> 4) * 8 * 32;
  if (gid >= total) return;
  int lane = gid & 31;
  int kt   = (gid >> 5) & 7;
  int t    = gid >> 8;
  int m    = lane & 15;
  int hl   = lane >> 4;
  long row = (long)(t * 16 + m);
  const float* p = A + row * 256 + kt * 32 + hl * 8;
  float s = rowscale ? rowscale[row] : 1.0f;
  float4 f0 = *(const float4*)(p);
  float4 f1 = *(const float4*)(p + 4);
  float4 f2 = *(const float4*)(p + 16);
  float4 f3 = *(const float4*)(p + 20);
  v16u u;
  u[0]  = f32_to_bf16(f0.x * s); u[1]  = f32_to_bf16(f0.y * s);
  u[2]  = f32_to_bf16(f0.z * s); u[3]  = f32_to_bf16(f0.w * s);
  u[4]  = f32_to_bf16(f1.x * s); u[5]  = f32_to_bf16(f1.y * s);
  u[6]  = f32_to_bf16(f1.z * s); u[7]  = f32_to_bf16(f1.w * s);
  u[8]  = f32_to_bf16(f2.x * s); u[9]  = f32_to_bf16(f2.y * s);
  u[10] = f32_to_bf16(f2.z * s); u[11] = f32_to_bf16(f2.w * s);
  u[12] = f32_to_bf16(f3.x * s); u[13] = f32_to_bf16(f3.y * s);
  u[14] = f32_to_bf16(f3.z * s); u[15] = f32_to_bf16(f3.w * s);
  *(v16u*)(pk + (long)gid * 16) = u;
}

// Layer 1: h1 = sigmoid(x @ Wself + aggmean @ Wneigh + b1).
// One wave owns one 16-row tile and 8 of the 16 column tiles: A-fragments are
// loaded ONCE per kt (32B each from the packed bf16 arrays) and reused across
// 8 column tiles; weight fragments stream from L2 (256KB total, always hot).
// 8 v8f accumulators = 64 VGPRs; 128 v_wmma per wave, fully unrolled.
__global__ void sage_layer1_wmma_kernel(const unsigned short* __restrict__ xpk,
                                        const unsigned short* __restrict__ gpk,
                                        const unsigned short* __restrict__ wpk,
                                        const float* __restrict__ b1,
                                        float* __restrict__ h1, int nNodes) {
  int wave = (blockIdx.x * blockDim.x + threadIdx.x) >> 5;
  int lane = threadIdx.x & 31;
  int nWaves = (nNodes >> 4) * 2;
  if (wave >= nWaves) return;
  int t   = wave >> 1;           // 16-row tile
  int cth = wave & 1;            // column-tile half: ct = cth*8 + j
  int m   = lane & 15;
  int hl  = lane >> 4;
  int node0 = t << 4;

  const unsigned short* pa_x = xpk + ((long)(t * 8) * 32 + lane) * 16;
  const unsigned short* pa_g = gpk + ((long)(t * 8) * 32 + lane) * 16;

  v8f acc[8] = {};
#pragma unroll
  for (int kt = 0; kt < 8; ++kt) {
    v16bf a_self = __builtin_bit_cast(v16bf, *(const v16u*)(pa_x + (long)kt * 32 * 16));
    v16bf a_ngh  = __builtin_bit_cast(v16bf, *(const v16u*)(pa_g + (long)kt * 32 * 16));
#pragma unroll
    for (int j = 0; j < 8; ++j) {
      int ct = cth * 8 + j;
      const unsigned short* pb = wpk + ((long)(ct * 8 + kt) * 32 + lane) * 16;
      v16bf b_self = __builtin_bit_cast(v16bf, *(const v16u*)(pb));
      v16bf b_ngh  = __builtin_bit_cast(v16bf, *(const v16u*)(pb + 65536));
      acc[j] = __builtin_amdgcn_wmma_f32_16x16x32_bf16(false, a_self, false, b_self,
                                                       (short)0, acc[j], false, false);
      acc[j] = __builtin_amdgcn_wmma_f32_16x16x32_bf16(false, a_ngh, false, b_ngh,
                                                       (short)0, acc[j], false, false);
    }
  }

  // D layout: VGPR r -> M = r + 8*hl, N = m
#pragma unroll
  for (int j = 0; j < 8; ++j) {
    int   col  = (cth * 8 + j) * 16 + m;
    float bias = b1[col];
#pragma unroll
    for (int r = 0; r < 8; ++r) {
      int   row = node0 + hl * 8 + r;
      float tt  = acc[j][r] + bias;
      h1[(long)row * 256 + col] = 1.0f / (1.0f + __expf(-tt));
    }
  }
}

// Layer 2: out[n] = b2 + h1[n,:]·Wself2 + aggmean2[n,:]·Wneigh2 ; one wave per node.
__global__ void sage_layer2_kernel(const float* __restrict__ h1, const float* __restrict__ agg,
                                   const float* __restrict__ ws2, const float* __restrict__ wn2,
                                   const float* __restrict__ b2, float* __restrict__ out, int nNodes) {
  int gid  = blockIdx.x * blockDim.x + threadIdx.x;
  int node = gid >> 5;
  int lane = gid & 31;
  if (node >= nNodes) return;
  long base = (long)node * 256 + lane * 8;
  const float4* ph = (const float4*)(h1 + base);
  const float4* pa = (const float4*)(agg + base);
  const float4* pw = (const float4*)(ws2 + lane * 8);
  const float4* pn = (const float4*)(wn2 + lane * 8);
  float s = 0.f;
#pragma unroll
  for (int c = 0; c < 2; ++c) {
    float4 a = ph[c], w = pw[c];
    s += a.x * w.x + a.y * w.y + a.z * w.z + a.w * w.w;
    float4 g = pa[c], n = pn[c];
    s += g.x * n.x + g.y * n.y + g.z * n.z + g.w * n.w;
  }
#pragma unroll
  for (int off = 16; off > 0; off >>= 1) s += __shfl_xor(s, off, 32);
  if (lane == 0) out[node] = s + b2[0];
}

extern "C" void kernel_launch(void* const* d_in, const int* in_sizes, int n_in,
                              void* d_out, int out_size, void* d_ws, size_t ws_size,
                              hipStream_t stream) {
  const float* x   = (const float*)d_in[0];
  const int*   src = (const int*)  d_in[1];
  const int*   dst = (const int*)  d_in[2];
  const float* Ws1 = (const float*)d_in[3];
  const float* Wn1 = (const float*)d_in[4];
  const float* b1  = (const float*)d_in[5];
  const float* Ws2 = (const float*)d_in[6];
  const float* Wn2 = (const float*)d_in[7];
  const float* b2  = (const float*)d_in[8];
  float* out = (float*)d_out;

  const int nNodes = in_sizes[0] / 256;
  const int nE     = in_sizes[1];

  // Workspace: agg[N*256]f32 | h1[N*256]f32 | invdeg[N]f32 | wpk[131072]u16
  //            | xpk[N*256]u16 | gpk[N*256]u16
  float* agg    = (float*)d_ws;
  float* h1     = agg + (long)nNodes * 256;
  float* invdeg = h1  + (long)nNodes * 256;
  size_t off    = (((size_t)2 * nNodes * 256 + nNodes) * sizeof(float) + 63) & ~(size_t)63;
  unsigned short* wpk = (unsigned short*)((char*)d_ws + off);
  unsigned short* xpk = wpk + 2 * 65536;
  unsigned short* gpk = xpk + (long)nNodes * 256;

  const int B = 256;
  long nAgg4 = (long)nNodes * 64;                        // float4 count of agg
  int  blkAgg4  = (int)((nAgg4 + B - 1) / B);
  int  blkDeg4  = (nNodes / 4 + B - 1) / B;
  int  blkNodeW = ((long)nNodes * 32 + B - 1) / B;       // wave-per-node kernels
  int  blkEdgeW = ((long)nE * 32 + B - 1) / B;           // wave-per-edge kernels
  int  nPackA   = (nNodes >> 4) * 8 * 32;
  int  blkPackA = (nPackA + B - 1) / B;
  int  blkL1    = (((nNodes >> 4) * 2) * 32 + B - 1) / B;

  // ---- degree + weight/x packing + agg(x) ----
  zero_f4_kernel<<<blkAgg4, B, 0, stream>>>(agg, nAgg4);
  zero_f4_kernel<<<blkDeg4, B, 0, stream>>>(invdeg, nNodes / 4);
  pack_w_kernel<<<(2 * 65536 + B - 1) / B, B, 0, stream>>>(Ws1, Wn1, wpk);
  pack_a_kernel<<<blkPackA, B, 0, stream>>>(x, (const float*)nullptr, xpk, nNodes);
  deg_accum_kernel<<<(nE + B - 1) / B, B, 0, stream>>>(dst, invdeg, nE);
  scatter_add_kernel<<<blkEdgeW, B, 0, stream>>>(x, src, dst, agg, nE);
  invdeg_kernel<<<(nNodes + B - 1) / B, B, 0, stream>>>(invdeg, nNodes);
  pack_a_kernel<<<blkPackA, B, 0, stream>>>(agg, invdeg, gpk, nNodes);  // fused mean + bf16

  // ---- layer 1 (WMMA bf16 -> f32, fused bias + sigmoid) ----
  sage_layer1_wmma_kernel<<<blkL1, B, 0, stream>>>(xpk, gpk, wpk, b1, h1, nNodes);

  // ---- aggregate h1, layer 2 ----
  zero_f4_kernel<<<blkAgg4, B, 0, stream>>>(agg, nAgg4);
  scatter_add_kernel<<<blkEdgeW, B, 0, stream>>>(h1, src, dst, agg, nE);
  scale_rows_kernel<<<blkAgg4, B, 0, stream>>>(agg, invdeg, nAgg4);
  sage_layer2_kernel<<<blkNodeW, B, 0, stream>>>(h1, agg, Ws2, Wn2, b2, out, nNodes);
}
